// GraphAE_82429012344989
// MI455X (gfx1250) — matrix operand order repaired and checked by reading, not compiled, not run on hardware
//
#include <hip/hip_runtime.h>
#include <hip/hip_bf16.h>
#include <math.h>

// ---------------- problem constants (from the reference) ----------------
#define NGRAPH 256
#define NNODE  50
#define NTOT   (NGRAPH * NNODE)      // 12800 nodes
#define DEG    49                    // fully connected, no self loops
#define EDGES  (NGRAPH * NNODE * DEG)
#define PI_F   3.14159265358979323846f

typedef __attribute__((ext_vector_type(16))) _Float16 v16h;
typedef __attribute__((ext_vector_type(8)))  float    v8f;

__device__ __forceinline__ float leakyf(float z) { return z >= 0.f ? z : 0.1f * z; }

// ============================================================================
// WMMA GEMM:  C[M x No] = A16[M x Kp] * Bt16[Np x Kp]^T   (Bt is N-major,
// zero-padded to Np = ceil(No/16)*16 rows so the K-loop has NO guards).
// 4 waves / block, one 16x16 tile per wave, sharing the same M-tile so the
// A fragments hit WGP$ across waves. epilogue: 0 none, 1 +bias, 2 leaky(+bias)
// Fragment layouts per CDNA5 ISA 7.12.2.
// ============================================================================
__global__ void wmma_gemm_f16(const _Float16* __restrict__ A,
                              const _Float16* __restrict__ Bt,
                              const float* __restrict__ bias,
                              float* __restrict__ C,
                              int Kp, int No, int ldc, int epilogue)
{
    const int wave = threadIdx.x >> 5;
    const int lane = threadIdx.x & 31;
    const int hf   = lane >> 4;              // 0: lanes 0-15, 1: lanes 16-31
    const int l16  = lane & 15;
    const int tn   = blockIdx.x * 4 + wave;  // N tile (one per wave)
    const int tm   = blockIdx.y;             // M tile
    const int ntiles = (No + 15) >> 4;
    if (tn >= ntiles) return;                // uniform per-wave exit (pre-WMMA)

    const int row = tm * 16 + l16;           // A row this lane feeds (M = lane%16)
    const int col = tn * 16 + l16;           // output column / Bt row (padded: always valid)

    v8f acc = {};
    const _Float16* arow = A  + (size_t)row * Kp;
    const _Float16* brow = Bt + (size_t)col * Kp;

    for (int k0 = 0; k0 < Kp; k0 += 32) {
        // A 16x32 f16: lanes0-15 hold K=[0..7]+[16..23], lanes16-31 K=[8..15]+[24..31]
        v16h a;
        const _Float16* ap = arow + k0 + 8 * hf;
#pragma unroll
        for (int i = 0; i < 8; ++i) { a[i] = ap[i]; a[8 + i] = ap[16 + i]; }
        // B 32x16 f16 (N-major source): lanes0-15 K=[0..7]+[16..23], lanes16-31 K=[8..15]+[24..31]
        v16h b;
        const _Float16* bp = brow + k0 + 8 * hf;
#pragma unroll
        for (int i = 0; i < 8; ++i) { b[i] = bp[i]; b[8 + i] = bp[16 + i]; }
        acc = __builtin_amdgcn_wmma_f32_16x16x32_f16(false, a, false, b,
                                                     (short)0, acc, false, false);
    }

    if (col < No) {
        float bv = (epilogue != 0) ? bias[col] : 0.f;
#pragma unroll
        for (int i = 0; i < 8; ++i) {
            int m = tm * 16 + i + 8 * hf;    // C/D: VGPR i -> M=i (lo half) / 8+i (hi half)
            float val = acc[i] + bv;
            if (epilogue == 2) val = leakyf(val);
            C[(size_t)m * ldc + col] = val;
        }
    }
}

// ---------------- f32 -> f16 with K padding (node features) ----------------
__global__ void to_f16_pad(const float* __restrict__ src, _Float16* __restrict__ dst,
                           int n, int Cin, int Kp)
{
    int t = blockIdx.x * blockDim.x + threadIdx.x;
    if (t >= n * Kp) return;
    int k = t % Kp, r = t / Kp;
    dst[t] = (k < Cin) ? (_Float16)src[(size_t)r * Cin + k] : (_Float16)0.f;
}

// EdgeConv weight prep: W[2*fi, fo] -> Pt[Np x Kp] = (W0-W1)^T, Qt[Np x Kp] = W1^T
// Rows o in [fo, Np) and cols k in [fi, Kp) are zero-padded.
__global__ void prep_conv_w(const float* __restrict__ W,
                            _Float16* __restrict__ Pt, _Float16* __restrict__ Qt,
                            int fi, int fo, int Kp, int Np)
{
    int t = blockIdx.x * blockDim.x + threadIdx.x;
    if (t >= Np * Kp) return;
    int k = t % Kp, o = t / Kp;
    if (k < fi && o < fo) {
        float w0 = W[(size_t)k * fo + o];
        float w1 = W[(size_t)(fi + k) * fo + o];
        Pt[t] = (_Float16)(w0 - w1);
        Qt[t] = (_Float16)w1;
    } else {
        Pt[t] = (_Float16)0.f; Qt[t] = (_Float16)0.f;
    }
}

// dec_fc2 weights [128, 6400] -> f16 transposed [6400 x 128] (6400 % 16 == 0)
__global__ void prep_fc2_w(const float* __restrict__ W, _Float16* __restrict__ Wt)
{
    int t = blockIdx.x * blockDim.x + threadIdx.x;
    if (t >= 6400 * 128) return;
    int k = t & 127, o = t >> 7;
    Wt[t] = (_Float16)W[(size_t)k * 6400 + o];
}

// e[:, 8:136] -> f16 [256 x 128]
__global__ void e_to_f16(const float* __restrict__ e, _Float16* __restrict__ out)
{
    int t = blockIdx.x * blockDim.x + threadIdx.x;
    if (t >= NGRAPH * 128) return;
    int k = t & 127, b = t >> 7;
    out[t] = (_Float16)e[(size_t)b * 136 + 8 + k];
}

// ============================================================================
// EdgeConv BatchNorm stats (closed form over fully connected graphs).
// Per-graph partials: part1 = sum_e h, part2 = sum_e h^2 contribution.
// ============================================================================
__global__ void conv_stats(const float* __restrict__ u, const float* __restrict__ v,
                           const float* __restrict__ bias,
                           float* __restrict__ part1, float* __restrict__ part2, int C)
{
    int g = blockIdx.x, c = threadIdx.x;
    if (c >= C) return;
    float b = bias[c];
    float Su = 0.f, Su2 = 0.f, Sv = 0.f, Sv2 = 0.f, Sd2 = 0.f;
    for (int i = 0; i < NNODE; ++i) {
        float uu = u[((size_t)g * NNODE + i) * C + c] + b;   // u' = u + bias
        float vv = v[((size_t)g * NNODE + i) * C + c];
        Su += uu; Su2 += uu * uu; Sv += vv; Sv2 += vv * vv;
        float d = uu + vv; Sd2 += d * d;
    }
    part1[(size_t)g * C + c] = (float)DEG * (Su + Sv);
    part2[(size_t)g * C + c] = (float)NNODE * (Su2 + Sv2) + 2.f * Su * Sv - Sd2;
}

// Deterministic fixed-order reduction over graphs -> BN scale/shift.
__global__ void conv_finalize(const float* __restrict__ part1, const float* __restrict__ part2,
                              const float* __restrict__ gamma, const float* __restrict__ beta,
                              const float* __restrict__ bias,
                              float* __restrict__ scale, float* __restrict__ shift, int C)
{
    int c = threadIdx.x;
    if (c >= C) return;
    float s1 = 0.f, s2 = 0.f;
    for (int g = 0; g < NGRAPH; ++g) {
        s1 += part1[(size_t)g * C + c];
        s2 += part2[(size_t)g * C + c];
    }
    const float invE = 1.0f / (float)EDGES;
    float mu  = s1 * invE;
    float var = s2 * invE - mu * mu;
    float sc  = gamma[c] * rsqrtf(var + 1e-5f);
    scale[c] = sc;
    shift[c] = sc * (bias[c] - mu) + beta[c];
}

// Pairwise leaky aggregation: out_d = (1/49) [ sum_s act(a_d + t_s) - act(a_d + t_d) ]
__global__ void conv_aggregate(const float* __restrict__ u, const float* __restrict__ v,
                               const float* __restrict__ scale, const float* __restrict__ shift,
                               float* __restrict__ out, int C, int act)
{
    __shared__ float t[NNODE * 128];
    int g = blockIdx.x, c = threadIdx.x;
    float sc = 0.f, sh = 0.f;
    if (c < C) {
        sc = scale[c]; sh = shift[c];
        for (int s = 0; s < NNODE; ++s)
            t[s * C + c] = sc * v[((size_t)g * NNODE + s) * C + c];
    }
    __syncthreads();
    if (c >= C) return;
    for (int d = 0; d < NNODE; ++d) {
        float a = sc * u[((size_t)g * NNODE + d) * C + c] + sh;
        float sum = 0.f;
        if (act) {
            for (int s = 0; s < NNODE; ++s) sum += leakyf(a + t[s * C + c]);
            sum -= leakyf(a + t[d * C + c]);
        } else {
            for (int s = 0; s < NNODE; ++s) sum += a + t[s * C + c];
            sum -= a + t[d * C + c];
        }
        out[((size_t)g * NNODE + d) * C + c] = sum * (1.0f / (float)DEG);
    }
}

// Per-graph mean/max pooling -> pooled[B x 256] = [mean(128) | max(128)]
__global__ void pool_graph(const float* __restrict__ h, float* __restrict__ pooled)
{
    int g = blockIdx.x, c = threadIdx.x;   // blockDim = 128
    float s = 0.f, m = -INFINITY;
    for (int i = 0; i < NNODE; ++i) {
        float x = h[((size_t)g * NNODE + i) * 128 + c];
        s += x; m = fmaxf(m, x);
    }
    pooled[(size_t)g * 256 + c]       = s * (1.0f / (float)NNODE);
    pooled[(size_t)g * 256 + 128 + c] = m;
}

// Generic small FC (f32): Y[r,o] = act(sum_k X[r,k]*W[k,o] + b[o]); strided rows.
__global__ void fc_f32(const float* __restrict__ X, int ldx,
                       const float* __restrict__ W, const float* __restrict__ bias,
                       float* __restrict__ Y, int ldy,
                       int rows, int K, int No, int act)
{
    int t = blockIdx.x * blockDim.x + threadIdx.x;
    if (t >= rows * No) return;
    int r = t / No, o = t % No;
    const float* x = X + (size_t)r * ldx;
    float s = bias[o];
    for (int k = 0; k < K; ++k) s += x[k] * W[(size_t)k * No + o];
    if (act) s = leakyf(s);
    Y[(size_t)r * ldy + o] = s;
}

// met head: e[:, :8] @ W[8,2] + b; relu col0, cycle col1 -> x_met_bar [B,2]
__global__ void met_head(const float* __restrict__ e, const float* __restrict__ W,
                         const float* __restrict__ b, float* __restrict__ out)
{
    int g = blockIdx.x * blockDim.x + threadIdx.x;
    if (g >= NGRAPH) return;
    const float* x = e + (size_t)g * 136;
    float m0 = b[0], m1 = b[1];
    for (int k = 0; k < 8; ++k) { m0 += x[k] * W[k * 2 + 0]; m1 += x[k] * W[k * 2 + 1]; }
    m0 = fmaxf(m0, 0.f);
    if (m1 >= PI_F)  m1 -= 2.f * PI_F;
    if (m1 < -PI_F)  m1 += 2.f * PI_F;
    out[g * 2 + 0] = m0; out[g * 2 + 1] = m1;
}

// take_along_axis: out[(b*50+n), c] = xxfc[b, idx[b,n]*128 + c]
__global__ void gather_nodes(const float* __restrict__ xxfc, const int* __restrict__ idx,
                             float* __restrict__ out)
{
    int t = blockIdx.x * blockDim.x + threadIdx.x;
    if (t >= NTOT * 128) return;
    int c = t & 127, node = t >> 7;
    int b = node / NNODE, n = node % NNODE;
    int j = idx[b * NNODE + n];
    out[t] = xxfc[(size_t)b * 6400 + (size_t)j * 128 + c];
}

// final node head: log_softmax(0:4) | relu(4:6) | 6*tanh(6) | 7*tanh(7)
__global__ void node_head(const float* __restrict__ h, float* __restrict__ xbar)
{
    int n = blockIdx.x * blockDim.x + threadIdx.x;
    if (n >= NTOT) return;
    const float* x = h + (size_t)n * 8;
    float m = fmaxf(fmaxf(x[0], x[1]), fmaxf(x[2], x[3]));
    float s = expf(x[0] - m) + expf(x[1] - m) + expf(x[2] - m) + expf(x[3] - m);
    float lse = m + logf(s);
    float* o = xbar + (size_t)n * 8;
    o[0] = x[0] - lse; o[1] = x[1] - lse; o[2] = x[2] - lse; o[3] = x[3] - lse;
    o[4] = fmaxf(x[4], 0.f); o[5] = fmaxf(x[5], 0.f);
    o[6] = 6.f * tanhf(x[6]); o[7] = 7.f * tanhf(x[7]);
}

// ============================================================================
extern "C" void kernel_launch(void* const* d_in, const int* in_sizes, int n_in,
                              void* d_out, int out_size, void* d_ws, size_t ws_size,
                              hipStream_t stream)
{
    (void)in_sizes; (void)n_in; (void)out_size; (void)ws_size;

    // ---- inputs (setup_inputs() order; pytrees flattened in insertion order) ----
    const float* x     = (const float*)d_in[0];   // [N,5]
    const float* x_met = (const float*)d_in[1];   // [B,2]
    const int*   idx   = (const int*)d_in[4];     // [B,50]
    struct ConvP { const float *W, *b, *g, *bt; };
    ConvP enc[3], dec[4];
    for (int i = 0; i < 3; ++i)
        enc[i] = { (const float*)d_in[5 + 4*i], (const float*)d_in[6 + 4*i],
                   (const float*)d_in[7 + 4*i], (const float*)d_in[8 + 4*i] };
    for (int i = 0; i < 4; ++i)
        dec[i] = { (const float*)d_in[17 + 4*i], (const float*)d_in[18 + 4*i],
                   (const float*)d_in[19 + 4*i], (const float*)d_in[20 + 4*i] };
    const float* W_encfc1  = (const float*)d_in[33]; const float* b_encfc1  = (const float*)d_in[34];
    const float* W_encmet  = (const float*)d_in[35]; const float* b_encmet  = (const float*)d_in[36];
    const float* W_encfc2  = (const float*)d_in[37]; const float* b_encfc2  = (const float*)d_in[38];
    const float* W_decfc1  = (const float*)d_in[39]; const float* b_decfc1  = (const float*)d_in[40];
    const float* W_decmet  = (const float*)d_in[41]; const float* b_decmet  = (const float*)d_in[42];
    const float* W_decfc2  = (const float*)d_in[43]; const float* b_decfc2  = (const float*)d_in[44];

    // ---- workspace carve-up ----
    char* p = (char*)d_ws;
    auto alloc = [&](size_t bytes) -> char* {
        char* r = p; p += (bytes + 255) & ~(size_t)255; return r;
    };
    float*     featA  = (float*)alloc((size_t)NTOT * 128 * 4);
    float*     featB  = (float*)alloc((size_t)NTOT * 128 * 4);
    float*     ubuf   = (float*)alloc((size_t)NTOT * 128 * 4);
    float*     vbuf   = (float*)alloc((size_t)NTOT * 128 * 4);
    _Float16*  xh     = (_Float16*)alloc((size_t)NTOT * 128 * 2);
    _Float16*  Pt     = (_Float16*)alloc(128 * 128 * 2);
    _Float16*  Qt     = (_Float16*)alloc(128 * 128 * 2);
    float*     part1  = (float*)alloc((size_t)NGRAPH * 128 * 4);
    float*     part2  = (float*)alloc((size_t)NGRAPH * 128 * 4);
    float*     scaleb = (float*)alloc(128 * 4);
    float*     shiftb = (float*)alloc(128 * 4);
    float*     pooled = (float*)alloc((size_t)NGRAPH * 256 * 4);
    float*     zcat   = (float*)alloc((size_t)NGRAPH * 136 * 4);
    float*     zbuf   = (float*)alloc((size_t)NGRAPH * 64 * 4);
    float*     ebuf   = (float*)alloc((size_t)NGRAPH * 136 * 4);
    _Float16*  ef16   = (_Float16*)alloc((size_t)NGRAPH * 128 * 2);
    _Float16*  wfc2t  = (_Float16*)alloc((size_t)6400 * 128 * 2);
    float*     xxfc   = (float*)alloc((size_t)NGRAPH * 6400 * 4);

    float* out_xbar = (float*)d_out;                       // [N,8]
    float* out_met  = (float*)d_out + (size_t)NTOT * 8;    // [B,2]
    float* out_z    = out_met + (size_t)NGRAPH * 2;        // [B,64]

    // ---- fused EdgeConv: u/v WMMA GEMMs + closed-form BN + pairwise aggregate ----
    auto run_conv = [&](const float* fin, float* fout, const ConvP& cp,
                        int fi, int fo, int act) {
        int Kp = (fi + 31) & ~31;               // K padded to WMMA depth
        int Np = (fo + 15) & ~15;               // N padded so GEMM loop is guard-free
        int nA = NTOT * Kp;
        to_f16_pad<<<(nA + 255) / 256, 256, 0, stream>>>(fin, xh, NTOT, fi, Kp);
        int nW = Np * Kp;
        prep_conv_w<<<(nW + 255) / 256, 256, 0, stream>>>(cp.W, Pt, Qt, fi, fo, Kp, Np);
        dim3 grid((Np / 16 + 3) / 4, NTOT / 16);
        wmma_gemm_f16<<<grid, 128, 0, stream>>>(xh, Pt, nullptr, ubuf, Kp, fo, fo, 0);
        wmma_gemm_f16<<<grid, 128, 0, stream>>>(xh, Qt, nullptr, vbuf, Kp, fo, fo, 0);
        int bdim = ((fo + 31) / 32) * 32;
        conv_stats<<<NGRAPH, bdim, 0, stream>>>(ubuf, vbuf, cp.b, part1, part2, fo);
        conv_finalize<<<1, bdim, 0, stream>>>(part1, part2, cp.g, cp.bt, cp.b,
                                              scaleb, shiftb, fo);
        conv_aggregate<<<NGRAPH, bdim, 0, stream>>>(ubuf, vbuf, scaleb, shiftb, fout, fo, act);
    };

    // ================= encoder =================
    run_conv(x,     featA, enc[0],   5,  64, 1);
    run_conv(featA, featB, enc[1],  64, 128, 1);
    run_conv(featB, featA, enc[2], 128, 128, 1);

    pool_graph<<<NGRAPH, 128, 0, stream>>>(featA, pooled);

    // xm -> zcat[:, 0:8];  g -> zcat[:, 8:136]
    fc_f32<<<(NGRAPH * 8   + 255) / 256, 256, 0, stream>>>(x_met, 2,  W_encmet, b_encmet,
                                                           zcat,     136, NGRAPH,   2,   8, 1);
    fc_f32<<<(NGRAPH * 128 + 255) / 256, 256, 0, stream>>>(pooled, 256, W_encfc1, b_encfc1,
                                                           zcat + 8, 136, NGRAPH, 256, 128, 1);
    // z = zcat @ enc_fc2
    fc_f32<<<(NGRAPH * 64  + 255) / 256, 256, 0, stream>>>(zcat, 136, W_encfc2, b_encfc2,
                                                           zbuf,  64, NGRAPH, 136,  64, 0);
    hipMemcpyAsync(out_z, zbuf, (size_t)NGRAPH * 64 * 4, hipMemcpyDeviceToDevice, stream);

    // ================= decoder =================
    fc_f32<<<(NGRAPH * 136 + 255) / 256, 256, 0, stream>>>(zbuf, 64, W_decfc1, b_decfc1,
                                                           ebuf, 136, NGRAPH, 64, 136, 1);
    met_head<<<(NGRAPH + 255) / 256, 256, 0, stream>>>(ebuf, W_decmet, b_decmet, out_met);

    // dec_fc2 (the big FC) via WMMA: [256 x 128] @ [128 x 6400], fused bias+leaky
    e_to_f16<<<(NGRAPH * 128 + 255) / 256, 256, 0, stream>>>(ebuf, ef16);
    prep_fc2_w<<<(6400 * 128 + 255) / 256, 256, 0, stream>>>(W_decfc2, wfc2t);
    wmma_gemm_f16<<<dim3(6400 / 16 / 4, NGRAPH / 16), 128, 0, stream>>>(ef16, wfc2t, b_decfc2,
                                                                        xxfc, 128, 6400,
                                                                        6400, 2);
    gather_nodes<<<(NTOT * 128 + 255) / 256, 256, 0, stream>>>(xxfc, idx, featA);

    run_conv(featA, featB, dec[0], 128, 128, 1);
    run_conv(featB, featA, dec[1], 128, 128, 1);
    run_conv(featA, featB, dec[2], 128,  64, 1);
    run_conv(featB, featA, dec[3],  64,   8, 0);   // last layer: identity activation

    node_head<<<(NTOT + 255) / 256, 256, 0, stream>>>(featA, out_xbar);
}
